// SimpleGemNet_2783138808358
// MI455X (gfx1250) — compile-verified, matrix operand order checked by reading, not run on hardware
//
#include <hip/hip_runtime.h>
#include <math.h>

// ---------------------------------------------------------------------------
// SimpleGemNet forward for MI455X (gfx1250), wave32, bf16 WMMA.
// N=20000 nodes, E=320000 edges, D=H=256, ED=64, L=4 layers, G=64 graphs.
// ---------------------------------------------------------------------------

#define GN_N  20000
#define GN_E  320000
#define GN_D  256
#define GN_H  256
#define GN_ED 64
#define GN_L  4
#define GN_G  64
#define W1LD  520   // padded stride for transposed msg_w1 (K=513 -> 520, keeps 4B align)

typedef __attribute__((ext_vector_type(16))) __bf16 v16bf;
typedef __attribute__((ext_vector_type(2)))  __bf16 bf16x2;
typedef __attribute__((ext_vector_type(8)))  float  v8f;

// ---------------------------------------------------------------------------
// Wave-level dual 16x16 WMMA MAC: two M-tiles (A0 rows 0-15, A1 rows 16-31)
// against one shared N-tile of B.  Each B fragment is loaded ONCE and used by
// two v_wmma_f32_16x16x32_bf16 (independent accumulators -> no D->A/B hazard).
// A: LDS row-major [16][ldA]; B: global transposed weight [16][ldB] (N x K).
// Fragment layouts per CDNA5 ISA 7.12.2.
// ---------------------------------------------------------------------------
__device__ __forceinline__ void wmma_bf16_mac2(v8f& c0, v8f& c1,
                                               const __bf16* A0, const __bf16* A1, int ldA,
                                               const __bf16* B, int ldB,
                                               int kTiles, int lane) {
    const int half = lane >> 4;
    const int row  = lane & 15;   // A: M row of tile; B: N column of tile
    for (int kt = 0; kt < kTiles; ++kt) {
        const int kb = kt * 32 + 8 * half;
        v16bf a0, a1, b;
#pragma unroll
        for (int v = 0; v < 8; ++v) {
            const int k0 = kb + ((v < 4) ? (2 * v) : (16 + 2 * (v - 4)));
            bf16x2 pb = *(const bf16x2*)(B  + row * ldB + k0);
            bf16x2 p0 = *(const bf16x2*)(A0 + row * ldA + k0);
            bf16x2 p1 = *(const bf16x2*)(A1 + row * ldA + k0);
            b[2 * v]  = pb.x; b[2 * v + 1]  = pb.y;
            a0[2 * v] = p0.x; a0[2 * v + 1] = p0.y;
            a1[2 * v] = p1.x; a1[2 * v + 1] = p1.y;
        }
        c0 = __builtin_amdgcn_wmma_f32_16x16x32_bf16(false, a0, false, b,
                                                     (short)0, c0, false, false);
        c1 = __builtin_amdgcn_wmma_f32_16x16x32_bf16(false, a1, false, b,
                                                     (short)0, c1, false, false);
    }
}

__device__ __forceinline__ float silu_f(float v) {
    return v / (1.0f + __expf(-v));
}

// ---------------------------------------------------------------------------
// Weight convert+transpose: w[K][Nw] f32 -> wt[Nw][ldOut] bf16 (wt[n*ldOut+k])
// ---------------------------------------------------------------------------
__global__ void gn_transpose_bf16(const float* __restrict__ w, __bf16* __restrict__ wt,
                                  int K, int Nw, int ldOut) {
    int idx = blockIdx.x * 256 + threadIdx.x;
    if (idx < K * Nw) {
        int k = idx / Nw, n = idx % Nw;
        wt[n * ldOut + k] = (__bf16)w[idx];
    }
}

__global__ void gn_zero_f32(float* p, int n) {
    int i = blockIdx.x * 256 + threadIdx.x;
    if (i < n) p[i] = 0.0f;
}

__global__ void gn_edge_dist(const float* __restrict__ pos, const int* __restrict__ ei,
                             float* __restrict__ dist) {
    int e = blockIdx.x * 256 + threadIdx.x;
    if (e < GN_E) {
        int s = ei[e], d = ei[GN_E + e];
        float dx = pos[s * 3 + 0] - pos[d * 3 + 0];
        float dy = pos[s * 3 + 1] - pos[d * 3 + 1];
        float dz = pos[s * 3 + 2] - pos[d * 3 + 2];
        dist[e] = sqrtf(dx * dx + dy * dy + dz * dz);
    }
}

// ---------------------------------------------------------------------------
// Node embed: x[32 nodes][256] f32 @ node_w_t(bf16 [256][256]) + b -> x_emb bf16
// 256 threads (8 wave32). Each wave owns 2 N-tiles; both M-tiles per K-sweep.
// ---------------------------------------------------------------------------
__global__ void __launch_bounds__(256)
gn_node_embed(const float* __restrict__ x, const __bf16* __restrict__ wt,
              const float* __restrict__ b, __bf16* __restrict__ x_emb) {
    __shared__ __bf16 As[32 * 256];
    const int b0 = blockIdx.x * 32;
    for (int i = threadIdx.x; i < 32 * 256; i += 256) {
        int m = i >> 8, k = i & 255;
        As[i] = (__bf16)x[(size_t)(b0 + m) * 256 + k];
    }
    __syncthreads();
    const int wave = threadIdx.x >> 5, lane = threadIdx.x & 31;
    const int n_ = lane & 15, half = lane >> 4;
    for (int nt = wave; nt < 16; nt += 8) {
        v8f c0 = {}, c1 = {};
        wmma_bf16_mac2(c0, c1, As, As + 16 * 256, 256, wt + nt * 16 * 256, 256, 8, lane);
        int n = nt * 16 + n_;
        float bias = b[n];
#pragma unroll
        for (int r = 0; r < 8; ++r) {
            int m = half * 8 + r;
            x_emb[(size_t)(b0 + m) * 256 + n]      = (__bf16)(c0[r] + bias);
            x_emb[(size_t)(b0 + 16 + m) * 256 + n] = (__bf16)(c1[r] + bias);
        }
    }
}

// ---------------------------------------------------------------------------
// Fused message MLP + edge embed + scatter-add. 32 edges per block.
//   m_in = [x_emb[dst] | x_emb[src] | dist]     (K=512 WMMA + rank-1 dist term)
//   m    = silu(m_in @ w1 + b1)
//   m    = silu(m @ w2 + b2) + (edge_attr @ we + be)
//   aggr[dst] += m     (atomic f32)
// ---------------------------------------------------------------------------
__global__ void __launch_bounds__(256)
gn_message_scatter(const __bf16* __restrict__ x_emb,
                   const float* __restrict__ edge_attr,
                   const float* __restrict__ dist,
                   const int* __restrict__ src_idx,
                   const int* __restrict__ dst_idx,
                   const __bf16* __restrict__ w1t, const float* __restrict__ b1,
                   const __bf16* __restrict__ w2t, const float* __restrict__ b2,
                   const __bf16* __restrict__ wet, const float* __restrict__ be,
                   float* __restrict__ aggr) {
    __shared__ __bf16 A1[32 * 512];   // gathered [dst | src] activations
    __shared__ __bf16 Ae[32 * 64];    // edge_attr tile
    __shared__ __bf16 M1[32 * 256];   // hidden message
    __shared__ float  distS[32];
    __shared__ int    dstS[32], srcS[32];

    const int e0 = blockIdx.x * 32;
    if (threadIdx.x < 32) {
        distS[threadIdx.x] = dist[e0 + threadIdx.x];
        dstS[threadIdx.x]  = dst_idx[e0 + threadIdx.x];
        srcS[threadIdx.x]  = src_idx[e0 + threadIdx.x];
    }
    __syncthreads();

    for (int i = threadIdx.x; i < 32 * 256; i += 256) {
        int m = i >> 8, k = i & 255;                 // per iter: same row, coalesced k
        A1[m * 512 + k]       = x_emb[(size_t)dstS[m] * 256 + k];
        A1[m * 512 + 256 + k] = x_emb[(size_t)srcS[m] * 256 + k];
    }
    for (int i = threadIdx.x; i < 32 * 64; i += 256) {
        int m = i >> 6, k = i & 63;
        Ae[i] = (__bf16)edge_attr[(size_t)(e0 + m) * 64 + k];
    }
    __syncthreads();

    const int wave = threadIdx.x >> 5, lane = threadIdx.x & 31;
    const int n_ = lane & 15, half = lane >> 4;

    // GEMM1 (K=512) + dist rank-1 + bias + silu -> M1 (bf16)
    for (int nt = wave; nt < 16; nt += 8) {
        v8f c0 = {}, c1 = {};
        wmma_bf16_mac2(c0, c1, A1, A1 + 16 * 512, 512,
                       w1t + (size_t)(nt * 16) * W1LD, W1LD, 16, lane);
        int n = nt * 16 + n_;
        float bias  = b1[n];
        float wlast = (float)w1t[(size_t)n * W1LD + 512];   // msg_w1 row 512 (dist col)
#pragma unroll
        for (int r = 0; r < 8; ++r) {
            int m = half * 8 + r;
            M1[m * 256 + n]        = (__bf16)silu_f(c0[r] + bias + distS[m] * wlast);
            M1[(16 + m) * 256 + n] = (__bf16)silu_f(c1[r] + bias + distS[16 + m] * wlast);
        }
    }
    __syncthreads();

    // GEMM2 (K=256) + silu, edge-embed GEMM (K=64), then scatter-add
    for (int nt = wave; nt < 16; nt += 8) {
        v8f c0 = {}, c1 = {}, ce0 = {}, ce1 = {};
        wmma_bf16_mac2(c0, c1, M1, M1 + 16 * 256, 256, w2t + nt * 16 * 256, 256, 8, lane);
        wmma_bf16_mac2(ce0, ce1, Ae, Ae + 16 * 64, 64, wet + nt * 16 * 64, 64, 2, lane);
        int n = nt * 16 + n_;
        float bias2 = b2[n], biase = be[n];
#pragma unroll
        for (int r = 0; r < 8; ++r) {
            int m = half * 8 + r;
            float v0 = silu_f(c0[r] + bias2) + (ce0[r] + biase);
            float v1 = silu_f(c1[r] + bias2) + (ce1[r] + biase);
            atomicAdd(&aggr[(size_t)dstS[m] * 256 + n], v0);
            atomicAdd(&aggr[(size_t)dstS[16 + m] * 256 + n], v1);
        }
    }
}

// ---------------------------------------------------------------------------
// Update MLP: x = silu(aggr @ u1 + ub1) @ u2 + ub2.  32 nodes/block, fused.
// ---------------------------------------------------------------------------
__global__ void __launch_bounds__(256)
gn_update_nodes(const float* __restrict__ aggr,
                const __bf16* __restrict__ u1t, const float* __restrict__ ub1,
                const __bf16* __restrict__ u2t, const float* __restrict__ ub2,
                float* __restrict__ x_out) {
    __shared__ __bf16 Ain[32 * 256];
    __shared__ __bf16 Hid[32 * 256];
    const int b0 = blockIdx.x * 32;
    for (int i = threadIdx.x; i < 32 * 256; i += 256) {
        int m = i >> 8, k = i & 255;
        Ain[i] = (__bf16)aggr[(size_t)(b0 + m) * 256 + k];
    }
    __syncthreads();
    const int wave = threadIdx.x >> 5, lane = threadIdx.x & 31;
    const int n_ = lane & 15, half = lane >> 4;

    for (int nt = wave; nt < 16; nt += 8) {
        v8f c0 = {}, c1 = {};
        wmma_bf16_mac2(c0, c1, Ain, Ain + 16 * 256, 256, u1t + nt * 16 * 256, 256, 8, lane);
        int n = nt * 16 + n_;
        float bias = ub1[n];
#pragma unroll
        for (int r = 0; r < 8; ++r) {
            int m = half * 8 + r;
            Hid[m * 256 + n]        = (__bf16)silu_f(c0[r] + bias);
            Hid[(16 + m) * 256 + n] = (__bf16)silu_f(c1[r] + bias);
        }
    }
    __syncthreads();
    for (int nt = wave; nt < 16; nt += 8) {
        v8f c0 = {}, c1 = {};
        wmma_bf16_mac2(c0, c1, Hid, Hid + 16 * 256, 256, u2t + nt * 16 * 256, 256, 8, lane);
        int n = nt * 16 + n_;
        float bias = ub2[n];
#pragma unroll
        for (int r = 0; r < 8; ++r) {
            int m = half * 8 + r;
            x_out[(size_t)(b0 + m) * 256 + n]      = c0[r] + bias;
            x_out[(size_t)(b0 + 16 + m) * 256 + n] = c1[r] + bias;
        }
    }
}

__global__ void gn_pool(const float* __restrict__ x, const int* __restrict__ batch,
                        float* __restrict__ pooled) {
    int idx = blockIdx.x * 256 + threadIdx.x;
    if (idx < GN_N * 256) {
        int row = idx >> 8, d = idx & 255;
        atomicAdd(&pooled[(size_t)batch[row] * 256 + d], x[idx]);
    }
}

__global__ void gn_head(const float* __restrict__ pooled,
                        const float* __restrict__ w1, const float* __restrict__ b1,
                        const float* __restrict__ w2, const float* __restrict__ b2,
                        float* __restrict__ out) {
    __shared__ float hid[256];
    int g = blockIdx.x, t = threadIdx.x;
    float s = b1[t];
    for (int k = 0; k < 256; ++k) s += pooled[g * 256 + k] * w1[k * 256 + t];
    hid[t] = silu_f(s) * w2[t];
    __syncthreads();
    for (int st = 128; st > 0; st >>= 1) {
        if (t < st) hid[t] += hid[t + st];
        __syncthreads();
    }
    if (t == 0) out[g] = hid[0] + b2[0];
}

// ---------------------------------------------------------------------------
extern "C" void kernel_launch(void* const* d_in, const int* in_sizes, int n_in,
                              void* d_out, int out_size, void* d_ws, size_t ws_size,
                              hipStream_t stream) {
    const float* x_in   = (const float*)d_in[0];
    const float* eattr  = (const float*)d_in[1];
    const float* pos    = (const float*)d_in[2];
    const int*   eidx   = (const int*)d_in[3];
    const int*   batch  = (const int*)d_in[4];
    const float* node_w = (const float*)d_in[5];
    const float* node_b = (const float*)d_in[6];
    const float* edge_w = (const float*)d_in[7];
    const float* edge_b = (const float*)d_in[8];
    const float* msg_w1 = (const float*)d_in[9];
    const float* msg_b1 = (const float*)d_in[10];
    const float* msg_w2 = (const float*)d_in[11];
    const float* msg_b2 = (const float*)d_in[12];
    const float* upd_w1 = (const float*)d_in[13];
    const float* upd_b1 = (const float*)d_in[14];
    const float* upd_w2 = (const float*)d_in[15];
    const float* upd_b2 = (const float*)d_in[16];
    const float* fc_w1  = (const float*)d_in[17];
    const float* fc_b1  = (const float*)d_in[18];
    const float* fc_w2  = (const float*)d_in[19];
    const float* fc_b2  = (const float*)d_in[20];

    char* ws = (char*)d_ws;
    size_t off = 0;
    auto carve = [&](size_t bytes) -> void* {
        void* p = ws + off;
        off += (bytes + 511) & ~(size_t)511;
        return p;
    };
    float*  dist   = (float*)carve((size_t)GN_E * 4);
    float*  xa     = (float*)carve((size_t)GN_N * GN_D * 4);
    float*  xb     = (float*)carve((size_t)GN_N * GN_D * 4);
    __bf16* xemb   = (__bf16*)carve((size_t)GN_N * GN_H * 2);
    float*  aggr   = (float*)carve((size_t)GN_N * GN_H * 4);
    float*  pooled = (float*)carve((size_t)GN_G * GN_D * 4);
    __bf16* nwt    = (__bf16*)carve((size_t)GN_L * GN_H * GN_D * 2);
    __bf16* ewt    = (__bf16*)carve((size_t)GN_L * GN_H * GN_ED * 2);
    __bf16* w1t    = (__bf16*)carve((size_t)GN_L * GN_H * W1LD * 2);
    __bf16* w2t    = (__bf16*)carve((size_t)GN_L * GN_H * GN_H * 2);
    __bf16* u1t    = (__bf16*)carve((size_t)GN_L * GN_H * GN_H * 2);
    __bf16* u2t    = (__bf16*)carve((size_t)GN_L * GN_D * GN_H * 2);

    auto cdiv = [](int n, int d) { return (n + d - 1) / d; };

    // Convert + transpose all layer weights to bf16 [N][K] layouts.
    for (int l = 0; l < GN_L; ++l) {
        gn_transpose_bf16<<<cdiv(256 * 256, 256), 256, 0, stream>>>(
            node_w + (size_t)l * 256 * 256, nwt + (size_t)l * 256 * 256, 256, 256, 256);
        gn_transpose_bf16<<<cdiv(64 * 256, 256), 256, 0, stream>>>(
            edge_w + (size_t)l * 64 * 256, ewt + (size_t)l * 256 * 64, 64, 256, 64);
        gn_transpose_bf16<<<cdiv(513 * 256, 256), 256, 0, stream>>>(
            msg_w1 + (size_t)l * 513 * 256, w1t + (size_t)l * 256 * W1LD, 513, 256, W1LD);
        gn_transpose_bf16<<<cdiv(256 * 256, 256), 256, 0, stream>>>(
            msg_w2 + (size_t)l * 256 * 256, w2t + (size_t)l * 256 * 256, 256, 256, 256);
        gn_transpose_bf16<<<cdiv(256 * 256, 256), 256, 0, stream>>>(
            upd_w1 + (size_t)l * 256 * 256, u1t + (size_t)l * 256 * 256, 256, 256, 256);
        gn_transpose_bf16<<<cdiv(256 * 256, 256), 256, 0, stream>>>(
            upd_w2 + (size_t)l * 256 * 256, u2t + (size_t)l * 256 * 256, 256, 256, 256);
    }

    gn_edge_dist<<<cdiv(GN_E, 256), 256, 0, stream>>>(pos, eidx, dist);

    const float* xcur = x_in;
    for (int l = 0; l < GN_L; ++l) {
        gn_node_embed<<<GN_N / 32, 256, 0, stream>>>(
            xcur, nwt + (size_t)l * 256 * 256, node_b + l * 256, xemb);
        gn_zero_f32<<<cdiv(GN_N * 256, 256), 256, 0, stream>>>(aggr, GN_N * 256);
        gn_message_scatter<<<GN_E / 32, 256, 0, stream>>>(
            xemb, eattr, dist, eidx, eidx + GN_E,
            w1t + (size_t)l * 256 * W1LD, msg_b1 + l * 256,
            w2t + (size_t)l * 256 * 256, msg_b2 + l * 256,
            ewt + (size_t)l * 256 * 64, edge_b + l * 256, aggr);
        float* xnext = (l & 1) ? xb : xa;
        gn_update_nodes<<<GN_N / 32, 256, 0, stream>>>(
            aggr, u1t + (size_t)l * 256 * 256, upd_b1 + l * 256,
            u2t + (size_t)l * 256 * 256, upd_b2 + l * 256, xnext);
        xcur = xnext;
    }

    gn_zero_f32<<<cdiv(GN_G * 256, 256), 256, 0, stream>>>(pooled, GN_G * 256);
    gn_pool<<<cdiv(GN_N * 256, 256), 256, 0, stream>>>(xcur, batch, pooled);
    gn_head<<<GN_G, 256, 0, stream>>>(pooled, fc_w1, fc_b1, fc_w2, fc_b2, (float*)d_out);
}
